// TransformerBlock_24309514895799
// MI455X (gfx1250) — matrix operand order repaired and checked by reading, not compiled, not run on hardware
//
#include <hip/hip_runtime.h>

// ---------------- CDNA5 WMMA types ----------------
typedef __attribute__((ext_vector_type(16))) __bf16 v16bf;
typedef __attribute__((ext_vector_type(8)))  float  v8f;

union FragBF { v16bf v; unsigned u[8]; unsigned short h[16]; };
union AccF   { v8f   v; float f[8]; };

__device__ __forceinline__ unsigned short f32_to_bf16(float f) {
  unsigned u = __float_as_uint(f);
  unsigned r = u + 0x7FFFu + ((u >> 16) & 1u);   // round to nearest even
  return (unsigned short)(r >> 16);
}

// A-fragment (16x32 bf16) dword j -> K base (contiguous: K = 2j + 16*half)
__device__ __forceinline__ int kbA(int j, int half) {
  return (j << 1) + (half << 4);
}
// B-fragment (32x16 bf16) dword j -> K base
__device__ __forceinline__ int kbB(int j, int half) {
  return (j << 1) + (half << 4);
}

// -------- gfx1250 async global->LDS copy (16B per lane, ASYNCcnt) --------
__device__ __forceinline__ void async_cp16(unsigned short* ldst,
                                           const unsigned short* gsrc) {
  unsigned lds = (unsigned)(size_t)ldst;               // low 32 bits = LDS addr
  unsigned long long ga = (unsigned long long)(size_t)gsrc;
  asm volatile("global_load_async_to_lds_b128 %0, %1, off"
               :: "v"(lds), "v"(ga) : "memory");
}
__device__ __forceinline__ void wait_async0() {
  asm volatile("s_wait_asynccnt 0x0" ::: "memory");
}

// ---------------- fp32 -> bf16 convert ----------------
__global__ void cvt_f32_bf16_kernel(const float* __restrict__ in,
                                    unsigned short* __restrict__ out, int n4) {
  int i = blockIdx.x * blockDim.x + threadIdx.x;
  int stride = gridDim.x * blockDim.x;
  for (; i < n4; i += stride) {
    float4 f = ((const float4*)in)[i];
    ushort4 o;
    o.x = f32_to_bf16(f.x); o.y = f32_to_bf16(f.y);
    o.z = f32_to_bf16(f.z); o.w = f32_to_bf16(f.w);
    ((ushort4*)out)[i] = o;
  }
}

// ---------------- RMSNorm -> bf16 ----------------
__global__ void rmsnorm_bf16_kernel(const float* __restrict__ x,
                                    const float* __restrict__ g,
                                    unsigned short* __restrict__ out, int D) {
  int row = blockIdx.x;
  const float* xr = x + (size_t)row * D;
  float s = 0.f;
  for (int i = threadIdx.x; i < D; i += blockDim.x) { float v = xr[i]; s += v * v; }
  #pragma unroll
  for (int o = 16; o > 0; o >>= 1) s += __shfl_xor(s, o, 32);
  __shared__ float red[8];
  int wid = threadIdx.x >> 5;
  if ((threadIdx.x & 31) == 0) red[wid] = s;
  __syncthreads();
  if (threadIdx.x == 0) {
    float t = 0.f;
    #pragma unroll
    for (int i = 0; i < 8; ++i) t += red[i];
    red[0] = t;
  }
  __syncthreads();
  float inv = 1.0f / sqrtf(red[0] / (float)D + 1e-5f);
  for (int i = threadIdx.x; i < D; i += blockDim.x)
    out[(size_t)row * D + i] = f32_to_bf16(xr[i] * inv * g[i]);
}

// ---------------- RoPE (+ cast) fp32 -> bf16 ----------------
__global__ void rope_cvt_kernel(const float* __restrict__ in,
                                const int* __restrict__ pos,
                                unsigned short* __restrict__ out,
                                int total_pairs, int applyRope) {
  int idx = blockIdx.x * blockDim.x + threadIdx.x;
  if (idx >= total_pairs) return;
  int r  = idx >> 9;          // row (b*2048+s)
  int c2 = idx & 511;         // pair in row
  int h  = c2 >> 5;
  int i  = c2 & 31;           // pair within head (d_head/2 = 32)
  size_t base = (size_t)r * 1024 + h * 64 + 2 * i;
  float xe = in[base], xo = in[base + 1];
  if (applyRope) {
    int s = r & 2047;
    float p = (float)pos[s];
    float invf = __expf(-((float)i * (1.0f / 32.0f)) * 9.210340371976184f);
    float fr = p * invf;
    float c = cosf(fr), sn = sinf(fr);
    float re = xe * c - xo * sn;
    float ro = xe * sn + xo * c;
    xe = re; xo = ro;
  }
  out[base]     = f32_to_bf16(xe);
  out[base + 1] = f32_to_bf16(xo);
}

// ---------------- WMMA bf16 GEMM: out = A(MxK) * W(NxK)^T ----------------
// Double-buffered async-LDS pipeline.
// mode 0: f32 store   1: f32 + residual   2: bf16 store
// mode 3: bf16 store of swish(aux)*C  (SwiGLU fuse; aux = x@w1^T, C = x@w3^T)
__global__ __launch_bounds__(128)
void gemm_wmma_bf16_kernel(const unsigned short* __restrict__ A,
                           const unsigned short* __restrict__ W,
                           float* __restrict__ outF,
                           unsigned short* __restrict__ outB,
                           const float* __restrict__ res,
                           const float* __restrict__ aux,
                           int M, int N, int K, int mode) {
  __shared__ unsigned short lA[2][64 * 32];
  __shared__ unsigned short lB[2][64 * 32];
  int m0 = blockIdx.y * 64, n0 = blockIdx.x * 64;
  int tid = threadIdx.x;
  int lane = tid & 31, wv = tid >> 5;
  int half = lane >> 4, ln = lane & 15;
  int wm = (wv >> 1) * 32, wn = (wv & 1) * 32;
  // staging geometry: 256 x 16B chunks per 64x32 tile, 128 threads, 2 chunks each
  int r0 = tid >> 2;              // rows r0 and r0+32, same col
  int c0 = (tid & 3) << 3;
  AccF acc[2][2] = {};

  auto stage = [&](int buf, int k0) {
    async_cp16(&lA[buf][r0 * 32 + c0],        &A[(size_t)(m0 + r0) * K + k0 + c0]);
    async_cp16(&lA[buf][(r0 + 32) * 32 + c0], &A[(size_t)(m0 + r0 + 32) * K + k0 + c0]);
    async_cp16(&lB[buf][r0 * 32 + c0],        &W[(size_t)(n0 + r0) * K + k0 + c0]);
    async_cp16(&lB[buf][(r0 + 32) * 32 + c0], &W[(size_t)(n0 + r0 + 32) * K + k0 + c0]);
  };

  stage(0, 0);
  wait_async0();
  __syncthreads();

  for (int k0 = 0; k0 < K; k0 += 32) {
    int buf = (k0 >> 5) & 1;
    if (k0 + 32 < K) stage(buf ^ 1, k0 + 32);   // prefetch next tile (async)

    FragBF af[2], bf[2];
    #pragma unroll
    for (int i = 0; i < 2; ++i)
      #pragma unroll
      for (int j = 0; j < 8; ++j)
        af[i].u[j] = *(const unsigned*)&lA[buf][(wm + i * 16 + ln) * 32 + kbA(j, half)];
    #pragma unroll
    for (int i = 0; i < 2; ++i)
      #pragma unroll
      for (int j = 0; j < 8; ++j)
        bf[i].u[j] = *(const unsigned*)&lB[buf][(wn + i * 16 + ln) * 32 + kbB(j, half)];

    #pragma unroll
    for (int i = 0; i < 2; ++i)
      #pragma unroll
      for (int j = 0; j < 2; ++j)
        acc[i][j].v = __builtin_amdgcn_wmma_f32_16x16x32_bf16(
            false, af[i].v, false, bf[j].v, (short)0, acc[i][j].v, false, false);

    wait_async0();      // our prefetch for k0+32 is in LDS
    __syncthreads();    // all waves done with buf / done staging buf^1
  }

  #pragma unroll
  for (int i = 0; i < 2; ++i)
    #pragma unroll
    for (int j = 0; j < 2; ++j)
      #pragma unroll
      for (int v = 0; v < 8; ++v) {
        int gm = m0 + wm + i * 16 + v + half * 8;
        int gn = n0 + wn + j * 16 + ln;
        size_t idx = (size_t)gm * N + gn;
        float val = acc[i][j].f[v];
        if (mode == 0)      outF[idx] = val;
        else if (mode == 1) outF[idx] = val + res[idx];
        else if (mode == 2) outB[idx] = f32_to_bf16(val);
        else {
          float a = aux[idx];
          float sw = a / (1.0f + __expf(-a));
          outB[idx] = f32_to_bf16(sw * val);
        }
      }
}

// ---------------- Flash attention (causal), WMMA bf16 ----------------
// grid (S/64, H, B), 128 threads (4 waves, 16 queries each)
__global__ __launch_bounds__(128)
void flash_attn_kernel(const unsigned short* __restrict__ Qb,
                       const unsigned short* __restrict__ Kb,
                       const unsigned short* __restrict__ Vb,
                       unsigned short* __restrict__ Ob) {
  const int S = 2048, D = 1024;
  __shared__ unsigned short lK[64 * 64];
  __shared__ unsigned short lV[64 * 64];
  __shared__ unsigned short lP[4 * 16 * 64];
  int q0 = blockIdx.x * 64;
  int h = blockIdx.y;
  int rb = blockIdx.z * S;
  int tid = threadIdx.x;
  int lane = tid & 31, wv = tid >> 5;
  int half = lane >> 4, ln = lane & 15;
  unsigned short* lPw = lP + wv * (16 * 64);
  // staging geometry: 512 x 16B chunks per 64x64 tile -> 4 chunks/thread
  int rr = tid >> 3;              // rows rr, rr+16, rr+32, rr+48; same col
  int cc = (tid & 7) << 3;

  // Q fragments for this wave's 16 queries (d-chunks 0,32)
  int qrow = q0 + wv * 16 + ln;
  const unsigned short* qptr = Qb + (size_t)(rb + qrow) * D + h * 64;
  FragBF qf[2];
  #pragma unroll
  for (int c = 0; c < 2; ++c)
    #pragma unroll
    for (int j = 0; j < 8; ++j)
      qf[c].u[j] = *(const unsigned*)&qptr[c * 32 + kbA(j, half)];

  AccF oacc[4] = {};
  float m_v[8], l_v[8];
  #pragma unroll
  for (int v = 0; v < 8; ++v) { m_v[v] = -1e30f; l_v[v] = 0.f; }

  for (int kt = 0; kt <= q0; kt += 64) {
    #pragma unroll
    for (int i = 0; i < 4; ++i) {
      int row = rr + 16 * i;
      size_t g = (size_t)(rb + kt + row) * D + h * 64 + cc;
      async_cp16(&lK[row * 64 + cc], &Kb[g]);
      async_cp16(&lV[row * 64 + cc], &Vb[g]);
    }
    wait_async0();
    __syncthreads();

    // S = Q * K^T  (B frag element (kk,n) = K[key=n][kk], contiguous in lK row)
    AccF sc[4] = {};
    #pragma unroll
    for (int j = 0; j < 4; ++j)
      #pragma unroll
      for (int dc = 0; dc < 2; ++dc) {
        FragBF kf;
        #pragma unroll
        for (int jj = 0; jj < 8; ++jj)
          kf.u[jj] = *(const unsigned*)&lK[(j * 16 + ln) * 64 + dc * 32 + kbB(jj, half)];
        sc[j].v = __builtin_amdgcn_wmma_f32_16x16x32_bf16(
            false, qf[dc].v, false, kf.v, (short)0, sc[j].v, false, false);
      }

    // online softmax per row (row = v + half*8, uniform across 16 lanes of half)
    #pragma unroll
    for (int v = 0; v < 8; ++v) {
      int qg = q0 + wv * 16 + v + half * 8;
      float mx = -1e30f;
      #pragma unroll
      for (int j = 0; j < 4; ++j) {
        int key = kt + j * 16 + ln;
        float s = sc[j].f[v] * 0.125f;           // 1/sqrt(64)
        s = (key <= qg) ? s : -1e30f;            // causal mask
        sc[j].f[v] = s;
        mx = fmaxf(mx, s);
      }
      #pragma unroll
      for (int o = 1; o < 16; o <<= 1) mx = fmaxf(mx, __shfl_xor(mx, o, 32));
      float mnew = fmaxf(m_v[v], mx);
      float alpha = __expf(m_v[v] - mnew);
      float rs = 0.f;
      #pragma unroll
      for (int j = 0; j < 4; ++j) {
        float p = __expf(sc[j].f[v] - mnew);
        sc[j].f[v] = p;
        rs += p;
      }
      #pragma unroll
      for (int o = 1; o < 16; o <<= 1) rs += __shfl_xor(rs, o, 32);
      l_v[v] = l_v[v] * alpha + rs;
      m_v[v] = mnew;
      #pragma unroll
      for (int j = 0; j < 4; ++j) oacc[j].f[v] *= alpha;
    }

    // transpose P (C-layout -> A-layout) through per-wave LDS slab
    #pragma unroll
    for (int j = 0; j < 4; ++j)
      #pragma unroll
      for (int v = 0; v < 8; ++v)
        lPw[(v + half * 8) * 64 + j * 16 + ln] = f32_to_bf16(sc[j].f[v]);
    __syncthreads();

    // O += P(16x32) * V(32x64)
    #pragma unroll
    for (int kc = 0; kc < 2; ++kc) {
      FragBF pf;
      #pragma unroll
      for (int jj = 0; jj < 8; ++jj)
        pf.u[jj] = *(const unsigned*)&lPw[ln * 64 + kc * 32 + kbA(jj, half)];
      #pragma unroll
      for (int j = 0; j < 4; ++j) {
        FragBF vf;
        #pragma unroll
        for (int e = 0; e < 16; ++e) {
          int kk = kc * 32 + e + (half << 4);    // B layout: K = e + half*16
          vf.h[e] = lV[kk * 64 + j * 16 + ln];
        }
        oacc[j].v = __builtin_amdgcn_wmma_f32_16x16x32_bf16(
            false, pf.v, false, vf.v, (short)0, oacc[j].v, false, false);
      }
    }
    __syncthreads();
  }

  #pragma unroll
  for (int v = 0; v < 8; ++v) {
    float inv = 1.0f / l_v[v];
    size_t grow = (size_t)(rb + q0 + wv * 16 + v + half * 8) * D + h * 64;
    #pragma unroll
    for (int j = 0; j < 4; ++j)
      Ob[grow + j * 16 + ln] = f32_to_bf16(oacc[j].f[v] * inv);
  }
}

// ---------------- host launch ----------------
extern "C" void kernel_launch(void* const* d_in, const int* in_sizes, int n_in,
                              void* d_out, int out_size, void* d_ws, size_t ws_size,
                              hipStream_t stream) {
  (void)in_sizes; (void)n_in; (void)out_size;
  const int Bc = 4, S = 2048, D = 1024, H = 16, FF = 4096;
  const int M = Bc * S; // 8192

  const float* x   = (const float*)d_in[0];
  const int*   pos = (const int*)d_in[1];
  const float* q_w = (const float*)d_in[2];
  const float* k_w = (const float*)d_in[3];
  const float* v_w = (const float*)d_in[4];
  const float* o_w = (const float*)d_in[5];
  const float* g1  = (const float*)d_in[6];
  const float* g2  = (const float*)d_in[7];
  const float* w1  = (const float*)d_in[8];
  const float* w2  = (const float*)d_in[9];
  const float* w3  = (const float*)d_in[10];
  float* out = (float*)d_out;

  char* ws = (char*)d_ws;
  size_t off = 0;
  auto alloc = [&](size_t bytes) -> void* {
    void* p = ws + off;
    off += (bytes + 255) & ~(size_t)255;
    return p;
  };

  unsigned short* XnB  = (unsigned short*)alloc((size_t)M * D * 2);
  unsigned short* OnB  = (unsigned short*)alloc((size_t)M * D * 2);
  unsigned short* WqB  = (unsigned short*)alloc((size_t)D * D * 2);
  unsigned short* WkB  = (unsigned short*)alloc((size_t)D * D * 2);
  unsigned short* WvB  = (unsigned short*)alloc((size_t)D * D * 2);
  unsigned short* WoB  = (unsigned short*)alloc((size_t)D * D * 2);
  unsigned short* W1B  = (unsigned short*)alloc((size_t)FF * D * 2);
  unsigned short* W3B  = (unsigned short*)alloc((size_t)FF * D * 2);
  unsigned short* W2B  = (unsigned short*)alloc((size_t)D * FF * 2);
  unsigned short* Qb   = (unsigned short*)alloc((size_t)M * D * 2);
  unsigned short* Kb   = (unsigned short*)alloc((size_t)M * D * 2);
  unsigned short* Vb   = (unsigned short*)alloc((size_t)M * D * 2);
  unsigned short* attnB= (unsigned short*)alloc((size_t)M * D * 2);
  float*          OF   = (float*)alloc((size_t)M * D * 4);
  float*          aF   = (float*)alloc((size_t)M * FF * 4);
  unsigned short* hB   = (unsigned short*)alloc((size_t)M * FF * 2);
  float* tmpF = aF;   // QKV f32 staging reuses aF (dead before w1 GEMM)
  if (off > ws_size) return;

  auto cvt = [&](const float* src, unsigned short* dst, int n) {
    cvt_f32_bf16_kernel<<<1024, 256, 0, stream>>>(src, dst, n / 4);
  };
  auto gemm = [&](const unsigned short* Ab, const unsigned short* Wb,
                  float* of, unsigned short* ob, const float* res,
                  const float* aux, int Mm, int Nn, int Kk, int mode) {
    dim3 g(Nn / 64, Mm / 64);
    gemm_wmma_bf16_kernel<<<g, 128, 0, stream>>>(Ab, Wb, of, ob, res, aux, Mm, Nn, Kk, mode);
  };

  // weights -> bf16
  cvt(q_w, WqB, D * D);  cvt(k_w, WkB, D * D);
  cvt(v_w, WvB, D * D);  cvt(o_w, WoB, D * D);
  cvt(w1,  W1B, FF * D); cvt(w3,  W3B, FF * D); cvt(w2, W2B, D * FF);

  // x_norm = rmsnorm(x, g1)
  rmsnorm_bf16_kernel<<<M, 256, 0, stream>>>(x, g1, XnB, D);

  const int ropeBlocks = (M * (D / 2)) / 256;
  // Q = rope(Xn @ q_w^T), K = rope(Xn @ k_w^T), V = Xn @ v_w^T
  gemm(XnB, WqB, tmpF, nullptr, nullptr, nullptr, M, D, D, 0);
  rope_cvt_kernel<<<ropeBlocks, 256, 0, stream>>>(tmpF, pos, Qb, M * (D / 2), 1);
  gemm(XnB, WkB, tmpF, nullptr, nullptr, nullptr, M, D, D, 0);
  rope_cvt_kernel<<<ropeBlocks, 256, 0, stream>>>(tmpF, pos, Kb, M * (D / 2), 1);
  gemm(XnB, WvB, tmpF, nullptr, nullptr, nullptr, M, D, D, 0);
  rope_cvt_kernel<<<ropeBlocks, 256, 0, stream>>>(tmpF, pos, Vb, M * (D / 2), 0);

  // causal flash attention
  dim3 ga(S / 64, H, Bc);
  flash_attn_kernel<<<ga, 128, 0, stream>>>(Qb, Kb, Vb, attnB);

  // O = x + attn @ o_w^T
  gemm(attnB, WoB, OF, nullptr, x, nullptr, M, D, D, 1);

  // On = rmsnorm(O, g2)
  rmsnorm_bf16_kernel<<<M, 256, 0, stream>>>(OF, g2, OnB, D);

  // a = On @ w1^T ; h = swish(a) * (On @ w3^T) ; out = O + h @ w2^T
  gemm(OnB, W1B, aF, nullptr, nullptr, nullptr, M, FF, D, 0);
  gemm(OnB, W3B, nullptr, hB, nullptr, aF, M, FF, D, 3);
  gemm(hB,  W2B, out, nullptr, OF, nullptr, M, D, FF, 1);
}
